// TopKMaxPooling_Fused_12369505812958
// MI455X (gfx1250) — matrix operand order repaired and checked by reading, not compiled, not run on hardware
//
#include <hip/hip_runtime.h>
#include <hip/hip_bf16.h>

// Problem constants (match reference)
#define BB   2
#define LL   4096
#define SS   128
#define HKV  4
#define GG   16
#define DD   128
#define TOPK 16
#define BLK  32
#define WIN  64

typedef __attribute__((ext_vector_type(16))) __bf16 v16bf;
typedef __attribute__((ext_vector_type(8)))  __bf16 v8bf;
typedef __attribute__((ext_vector_type(8)))  float  v8f;
typedef __attribute__((ext_vector_type(4)))  int    v4i;

#define KSTRIDE 136          // padded bf16 elems per K row in LDS (272 B, bank-skewed)
#define WAVES   8            // waves per block (wave32 -> 256 threads)
#define QPW     8            // queries per wave

// CDNA5 async global->LDS staging, with synchronous fallback
#if defined(__has_builtin)
#if __has_builtin(__builtin_amdgcn_global_load_async_to_lds_b128) && \
    __has_builtin(__builtin_amdgcn_s_wait_asynccnt)
#define HAVE_ASYNC_LDS 1
#endif
#endif
#ifndef HAVE_ASYNC_LDS
#define HAVE_ASYNC_LDS 0
#endif

typedef __attribute__((address_space(1))) v4i* g_v4i_ptr;
typedef __attribute__((address_space(3))) v4i* l_v4i_ptr;

__global__ __launch_bounds__(256) void nsa_topk_wmma_kernel(
    const __bf16* __restrict__ Q,
    const __bf16* __restrict__ K,
    float* __restrict__ outScores,   // [B,L,HKV,G,TOPK] fp32
    int*   __restrict__ outIdx)      // [B,L,HKV,TOPK] int32
{
  __shared__ __align__(16) __bf16 sK[SS * KSTRIDE];

  const int tid  = threadIdx.x;
  const int lane = tid & 31;
  const int wave = tid >> 5;
  const int lrow = lane & 15;

  // gid -> (b, h, l-base); 64 consecutive queries per block
  const int nLB = LL / (WAVES * QPW);            // 64
  int gid = blockIdx.x;
  const int lb = gid % nLB;
  const int bh = gid / nLB;
  const int h  = bh % HKV;
  const int b  = bh / HKV;
  const int lBase = lb * (WAVES * QPW);

  // ---- stage K[b, :, h, :] (128 x 128 bf16 = 32 KiB) into LDS, padded rows ----
  {
    const int chunks = SS * (DD / 8);            // 16-byte chunks: 2048
#if HAVE_ASYNC_LDS
    for (int c = tid; c < chunks; c += 256) {
      const int s = c >> 4, col = c & 15;
      g_v4i_ptr src = (g_v4i_ptr)(
          K + (((size_t)b * SS + s) * HKV + h) * DD + col * 8);
      l_v4i_ptr dst = (l_v4i_ptr)(sK + s * KSTRIDE + col * 8);
      __builtin_amdgcn_global_load_async_to_lds_b128(src, dst, 0, 0);
    }
    __builtin_amdgcn_s_wait_asynccnt(0);
#else
    for (int c = tid; c < chunks; c += 256) {
      const int s = c >> 4, col = c & 15;
      const float4 v = *reinterpret_cast<const float4*>(
          K + (((size_t)b * SS + s) * HKV + h) * DD + col * 8);
      *reinterpret_cast<float4*>(sK + s * KSTRIDE + col * 8) = v;
    }
#endif
  }
  __syncthreads();

  const int aoff = (lane < 16) ? 0 : 16;         // A-fragment K-offset per half-wave
  const int boff = (lane < 16) ? 0 : 16;         // B-fragment K-offset per half-wave
  const float scale = 0.08838834764831845f;      // 1/sqrt(128)

  #pragma unroll 1
  for (int q = 0; q < QPW; ++q) {
    const int l = lBase + q * WAVES + wave;

    // prefetch next query's 4 KiB Q tile (32 lanes x 128 B) into cache;
    // no VGPR cost, no LOADcnt — hides HBM latency across q iterations
    if (q + 1 < QPW) {
      const __bf16* nextTile =
          Q + ((((size_t)b * LL + (l + WAVES)) * HKV + h) * GG) * DD;
      __builtin_prefetch(nextTile + (size_t)lane * 64, 0, 0);
    }

    // ---- load A fragments: Q[b,l,h, g=lrow, :] in 16x32 bf16 WMMA layout ----
    const __bf16* qrow =
        Q + ((((size_t)b * LL + l) * HKV + h) * GG + lrow) * DD;
    v16bf a[4];
    #pragma unroll
    for (int ks = 0; ks < 4; ++ks) {
      v8bf lo = *reinterpret_cast<const v8bf*>(qrow + ks * 32 + aoff);
      v8bf hi = *reinterpret_cast<const v8bf*>(qrow + ks * 32 + aoff + 8);
      a[ks] = __builtin_shufflevector(lo, hi,
                0,1,2,3,4,5,6,7,8,9,10,11,12,13,14,15);
    }

    // causal block limit: floor((l - WIN + 1) / BLK); blocks n >= limit masked
    int limit; {
      const int num = l - (WIN - 1);
      limit = (num >= 0) ? (num / BLK) : -((-num + BLK - 1) / BLK);
    }

    // ---- scores (8 tiles of 16x16) + group-max pool ----
    float pv[8];
    #pragma unroll
    for (int t = 0; t < 8; ++t) {
      v8f acc = {};
      #pragma unroll
      for (int ks = 0; ks < 4; ++ks) {
        const __bf16* bp = sK + (t * 16 + lrow) * KSTRIDE + ks * 32 + boff;
        v8bf lo = *reinterpret_cast<const v8bf*>(bp);
        v8bf hi = *reinterpret_cast<const v8bf*>(bp + 8);
        v16bf bf = __builtin_shufflevector(lo, hi,
                    0,1,2,3,4,5,6,7,8,9,10,11,12,13,14,15);
        acc = __builtin_amdgcn_wmma_f32_16x16x32_bf16(
                  false, a[ks], false, bf, (short)0, acc, false, false);
      }
      float m = acc[0];
      #pragma unroll
      for (int r = 1; r < 8; ++r) m = fmaxf(m, acc[r]);
      m = fmaxf(m, __shfl_xor(m, 16, 32));       // combine M=0..7 with M=8..15
      const int n = t * 16 + lrow;
      pv[t] = (n >= limit) ? -__builtin_inff() : m;
    }

    // ---- iterative top-16 (lower-index tie-break, like lax.top_k) ----
    int mySel = -1;                              // lane k (k<16) records k-th pick
    for (int k = 0; k < TOPK; ++k) {
      float v = pv[0]; int vi = lrow;
      #pragma unroll
      for (int t = 1; t < 8; ++t) {
        if (pv[t] > v) { v = pv[t]; vi = t * 16 + lrow; }
      }
      #pragma unroll
      for (int m2 = 1; m2 < 32; m2 <<= 1) {
        const float ov = __shfl_xor(v, m2, 32);
        const int   oi = __shfl_xor(vi, m2, 32);
        if (ov > v || (ov == v && oi < vi)) { v = ov; vi = oi; }
      }
      const int sel = (v == -__builtin_inff()) ? -1 : vi;
      if (lane == k) mySel = sel;
      if ((vi & 15) == lrow) {                   // invalidate winner (both copies)
        const int tw = vi >> 4;
        #pragma unroll
        for (int t = 0; t < 8; ++t) if (t == tw) pv[t] = -__builtin_inff();
      }
    }

    // ---- stable ascending sort of the 16 picked indices (-1 keyed as S) ----
    const int key = (lane < 16) ? ((mySel < 0) ? SS : mySel) : (1000 + lane);
    int rank = 0;
    #pragma unroll 1
    for (int j = 0; j < 16; ++j) {
      const int kj = __shfl(key, j, 32);
      rank += (kj < key || (kj == key && j < lane)) ? 1 : 0;
    }
    int idxSorted = -1;                          // lanes 0..15: sorted index k=lane
    #pragma unroll 1
    for (int j = 0; j < 16; ++j) {
      const int rj = __shfl(rank, j, 32);
      const int sj = __shfl(mySel, j, 32);
      if (rj == lane) idxSorted = sj;
    }

    // ---- re-score selected blocks: gather B columns from LDS, 4 WMMAs ----
    const int selK = __shfl(idxSorted, lrow, 32);   // column n=lrow's block id
    const int srow = (selK < 0) ? 0 : selK;
    v8f acc2 = {};
    #pragma unroll
    for (int ks = 0; ks < 4; ++ks) {
      const __bf16* bp = sK + srow * KSTRIDE + ks * 32 + boff;
      v8bf lo = *reinterpret_cast<const v8bf*>(bp);
      v8bf hi = *reinterpret_cast<const v8bf*>(bp + 8);
      v16bf bf = __builtin_shufflevector(lo, hi,
                  0,1,2,3,4,5,6,7,8,9,10,11,12,13,14,15);
      acc2 = __builtin_amdgcn_wmma_f32_16x16x32_bf16(
                 false, a[ks], false, bf, (short)0, acc2, false, false);
    }

    // ---- write out: [G,TOPK] fp32 tile + sorted indices ----
    const size_t obase = ((((size_t)b * LL + l) * HKV + h)) * (GG * TOPK);
    const bool valid = (selK >= 0);
    #pragma unroll
    for (int r = 0; r < 8; ++r) {
      const int g = (lane < 16) ? r : (8 + r);
      const float val = valid ? acc2[r] * scale : -__builtin_inff();
      outScores[obase + g * TOPK + lrow] = val;
    }
    if (lane < 16) {
      outIdx[(((size_t)b * LL + l) * HKV + h) * TOPK + lane] = idxSorted;
    }
  }
}

extern "C" void kernel_launch(void* const* d_in, const int* in_sizes, int n_in,
                              void* d_out, int out_size, void* d_ws, size_t ws_size,
                              hipStream_t stream) {
  (void)in_sizes; (void)n_in; (void)out_size; (void)d_ws; (void)ws_size;
  const __bf16* Q = reinterpret_cast<const __bf16*>(d_in[0]);
  const __bf16* K = reinterpret_cast<const __bf16*>(d_in[1]);
  float* outScores = reinterpret_cast<float*>(d_out);
  int*   outIdx    = reinterpret_cast<int*>(
      outScores + (size_t)BB * LL * HKV * GG * TOPK);

  const int nBlocks = BB * HKV * (LL / (WAVES * QPW));   // 512
  nsa_topk_wmma_kernel<<<nBlocks, 256, 0, stream>>>(Q, K, outScores, outIdx);
}
